// BiDACPI_69011534512279
// MI455X (gfx1250) — matrix operand order repaired and checked by reading, compiled, not run on hardware
//
#include <hip/hip_runtime.h>
#include <hip/hip_bf16.h>

typedef __attribute__((ext_vector_type(16))) _Float16 v16h;
typedef __attribute__((ext_vector_type(8)))  float    v8f;

#define ALPHA_LK 0.2f
#define NEGBIG   (-9.0e15f)

__device__ __forceinline__ float leaky_(float x) { return x > 0.f ? x : ALPHA_LK * x; }
__device__ __forceinline__ float elu_(float x)   { return x > 0.f ? x : (__expf(x) - 1.f); }

// ---------------------------------------------------------------------------
// Generic WMMA GEMM: C[M,N] = A[M,K] @ op(B) (+bias) with epilogue.
// A: f32 row-major (lda). B: f32 row-major; TRANS_B=0 -> (K,N) ldb, =1 -> (N,K) ldb.
// Batched via blockIdx.z with element strides sA/sB/sC.
//
// Block = 128 threads = 4 waves; wave w owns the 16x16 tile (blockIdx.y*4+w, blockIdx.x).
// Per 32-k chunk the block cooperatively stages the 32x16 B panel into LDS
// (coalesced float4 global reads -> f16 -> LDS in WMMA B-fragment order), double
// buffered, so each wave's B fragment is one contiguous 32-byte LDS read.
// A fragments are read directly from global as 4x float4 per lane (one 16-k run per lane, k = 16*half + s).
//
// Requires: M % 64 == 0, N % 16 == 0, K % 32 == 0 (all call sites satisfy this).
// epi: 0=none, 1=leaky_relu, 2=elu
// ---------------------------------------------------------------------------
template <int TRANS_B>
__global__ void wmma_gemm_kernel(const float* __restrict__ A,
                                 const float* __restrict__ B,
                                 const float* __restrict__ bias,
                                 float* __restrict__ C,
                                 int M, int Nc, int K,
                                 int lda, int ldb, int ldc,
                                 long sA, long sB, long sC,
                                 int epi)
{
    __shared__ __attribute__((aligned(32))) _Float16 bsh[2][512]; // [buf][n*16 + (k>>4)*256 + (k&15)]

    const int tid   = threadIdx.x;
    const int lane  = tid & 31;
    const int wave  = tid >> 5;
    const int mtile = blockIdx.y * 4 + wave;
    const int ntile = blockIdx.x;
    const int nbase = ntile * 16;

    A += (long)blockIdx.z * sA;
    B += (long)blockIdx.z * sB;
    C += (long)blockIdx.z * sC;

    const int ml   = lane & 15;
    const int half = lane >> 4;
    const int m0   = mtile * 16 + ml;     // A-fragment row
    const int n0   = nbase + ml;          // B-fragment / C column

    // cooperative stage of one 32x16 B chunk (k = kb..kb+31) into bsh[buf]
    auto stage = [&](int buf, int kb) {
        if (TRANS_B) {
            // B is (N,K): element(k,n) = B[(nbase+n)*ldb + kb + k]; contiguous along k
            const int n  = tid >> 3;            // 0..15
            const int k0 = (tid & 7) * 4;       // 0,4,...,28
            const float4 f = *(const float4*)&B[(long)(nbase + n) * ldb + kb + k0];
            _Float16* d = &bsh[buf][n * 16 + (k0 >> 4) * 256 + (k0 & 15)];
            d[0] = (_Float16)f.x; d[1] = (_Float16)f.y;
            d[2] = (_Float16)f.z; d[3] = (_Float16)f.w;
        } else {
            // B is (K,N): row kb+k contiguous along n
            const int k = tid >> 2;             // 0..31
            const int q = (tid & 3) * 4;        // 0,4,8,12
            const float4 f = *(const float4*)&B[(long)(kb + k) * ldb + nbase + q];
            const int kg = (k >> 4) * 256 + (k & 15);
            bsh[buf][(q + 0) * 16 + kg] = (_Float16)f.x;
            bsh[buf][(q + 1) * 16 + kg] = (_Float16)f.y;
            bsh[buf][(q + 2) * 16 + kg] = (_Float16)f.z;
            bsh[buf][(q + 3) * 16 + kg] = (_Float16)f.w;
        }
    };

    stage(0, 0);
    __syncthreads();

    v8f acc = {};
    for (int kb = 0; kb < K; kb += 32) {
        const int cur = (kb >> 5) & 1;
        if (kb + 32 < K) stage(cur ^ 1, kb + 32);   // overlap next chunk with WMMA

        // A 16x32 f16 fragment: per-lane one run of 16 consecutive k (k = 16*half + s)
        const float* arow = A + (long)m0 * lda + kb;
        if (kb + 64 < K) __builtin_prefetch(arow + 64, 0, 3);   // global_prefetch_b8
        const float4 fa0 = *(const float4*)&arow[16 * half];
        const float4 fa1 = *(const float4*)&arow[16 * half + 4];
        const float4 fa2 = *(const float4*)&arow[16 * half + 8];
        const float4 fa3 = *(const float4*)&arow[16 * half + 12];
        v16h af;
        af[0]  = (_Float16)fa0.x; af[1]  = (_Float16)fa0.y;
        af[2]  = (_Float16)fa0.z; af[3]  = (_Float16)fa0.w;
        af[4]  = (_Float16)fa1.x; af[5]  = (_Float16)fa1.y;
        af[6]  = (_Float16)fa1.z; af[7]  = (_Float16)fa1.w;
        af[8]  = (_Float16)fa2.x; af[9]  = (_Float16)fa2.y;
        af[10] = (_Float16)fa2.z; af[11] = (_Float16)fa2.w;
        af[12] = (_Float16)fa3.x; af[13] = (_Float16)fa3.y;
        af[14] = (_Float16)fa3.z; af[15] = (_Float16)fa3.w;

        // B fragment: one contiguous 32B LDS read per lane
        const v16h bf = *(const v16h*)&bsh[cur][lane * 16];

        acc = __builtin_amdgcn_wmma_f32_16x16x32_f16(
                  false, af, false, bf, (short)0, acc, false, false);
        __syncthreads();
    }

    const float bb = bias ? bias[n0] : 0.f;
    #pragma unroll
    for (int r = 0; r < 8; ++r) {
        const int m = mtile * 16 + r + 8 * half;   // C/D f32 layout
        float v = acc[r] + bb;
        if      (epi == 1) v = leaky_(v);
        else if (epi == 2) v = elu_(v);
        C[(long)m * ldc + n0] = v;
    }
}

// ---------------------------------------------------------------------------
__global__ void gather_kernel(const int* __restrict__ idx, const float* __restrict__ E,
                              float* __restrict__ out, long total, int C)
{
    long t = (long)blockIdx.x * blockDim.x + threadIdx.x;
    if (t >= total) return;
    long r = t / C; int c = (int)(t - r * C);
    out[t] = E[(long)idx[r] * C + c];
}

// src/dst attention logits: one thread per graph node row
__global__ void srcdst_kernel(const float* __restrict__ Wh, const float* __restrict__ avec,
                              int F, int rows, float* __restrict__ src, float* __restrict__ dst)
{
    int r = blockIdx.x * blockDim.x + threadIdx.x;
    if (r >= rows) return;
    const float* w = Wh + (long)r * F;
    float s = 0.f, d = 0.f;
    for (int f = 0; f < F; ++f) { s += w[f] * avec[f]; d += w[f] * avec[F + f]; }
    src[r] = s; dst[r] = d;
}

// Masked row softmax over N=512 columns; 256 threads, 2 cols/thread, LDS reductions
__global__ void attn_softmax_kernel(const float* __restrict__ src, const float* __restrict__ dst,
                                    const int* __restrict__ adj, float* __restrict__ att, int Nn)
{
    const int row = blockIdx.x;          // b*N + i
    const int b   = row / Nn;
    const int t   = threadIdx.x;
    __shared__ float red[256];

    const float  si   = src[row];
    const float* drow = dst + (long)b * Nn;
    const int*   arow = adj + (long)row * Nn;
    const int j0 = t, j1 = t + 256;

    float v0 = leaky_(si + drow[j0]);
    float e0 = (arow[j0] > 0) ? v0 : NEGBIG;
    float v1 = leaky_(si + drow[j1]);
    float e1 = (arow[j1] > 0) ? v1 : NEGBIG;

    red[t] = fmaxf(e0, e1); __syncthreads();
    for (int s = 128; s > 0; s >>= 1) {
        if (t < s) red[t] = fmaxf(red[t], red[t + s]);
        __syncthreads();
    }
    const float mx = red[0]; __syncthreads();

    const float x0 = __expf(e0 - mx), x1 = __expf(e1 - mx);
    red[t] = x0 + x1; __syncthreads();
    for (int s = 128; s > 0; s >>= 1) {
        if (t < s) red[t] += red[t + s];
        __syncthreads();
    }
    const float inv = 1.f / red[0];

    float* orow = att + (long)row * Nn;
    orow[j0] = x0 * inv;
    orow[j1] = x1 * inv;
}

// 11x11 SAME conv over (L, W) "image", 1 channel, + bias + relu
__global__ void conv11_kernel(const float* __restrict__ in, const float* __restrict__ w,
                              const float* __restrict__ bias, float* __restrict__ out,
                              int L, int W, int Bn)
{
    long t = (long)blockIdx.x * blockDim.x + threadIdx.x;
    const long plane = (long)L * W;
    if (t >= (long)Bn * plane) return;
    const int b = (int)(t / plane);
    const long rem = t - (long)b * plane;
    const int y = (int)(rem / W), xc = (int)(rem % W);
    const float* img = in + (long)b * plane;

    float s = bias[0];
    #pragma unroll
    for (int ky = 0; ky < 11; ++ky) {
        const int yy = y + ky - 5;
        if (yy < 0 || yy >= L) continue;
        const float* irow = img + (long)yy * W;
        #pragma unroll
        for (int kx = 0; kx < 11; ++kx) {
            const int xx = xc + kx - 5;
            if (xx < 0 || xx >= W) continue;
            s += irow[xx] * w[ky * 11 + kx];
        }
    }
    out[t] = fmaxf(s, 0.f);
}

// Masked mean pool of a_v (over N) and p_v (over L), concat, leaky x2
__global__ void pool_cp_kernel(const float* __restrict__ a_v, const float* __restrict__ p_v,
                               const float* __restrict__ amask, const float* __restrict__ pmask,
                               float* __restrict__ cp, int N, int L, int F)
{
    const int b = blockIdx.x, t = threadIdx.x;   // 2F threads
    float acc = 0.f, msum = 0.f;
    if (t < F) {
        const float* base = a_v + (long)b * N * F + t;
        const float* mk = amask + (long)b * N;
        for (int n = 0; n < N; ++n) { acc += base[(long)n * F] * mk[n]; msum += mk[n]; }
    } else {
        const int f = t - F;
        const float* base = p_v + (long)b * L * F + f;
        const float* mk = pmask + (long)b * L;
        for (int n = 0; n < L; ++n) { acc += base[(long)n * F] * mk[n]; msum += mk[n]; }
    }
    float v = acc / msum;
    v = leaky_(v); v = leaky_(v);
    cp[b * 2 * F + t] = v;
}

__global__ void pred_kernel(const float* __restrict__ cp, const float* __restrict__ pw,
                            const float* __restrict__ pb, float* __restrict__ out, int D)
{
    const int b = blockIdx.x, t = threadIdx.x;
    __shared__ float red[256];
    red[t] = cp[b * D + t] * pw[t];
    __syncthreads();
    for (int s = 128; s > 0; s >>= 1) {
        if (t < s) red[t] += red[t + s];
        __syncthreads();
    }
    if (t == 0) out[b] = red[0] + pb[0];
}

// ---------------------------------------------------------------------------
template <int TB>
static inline void launch_gemm(const float* A, const float* B, const float* bias, float* C,
                               int M, int Nc, int K, int lda, int ldb, int ldc,
                               long sA, long sB, long sC, int batch, int epi,
                               hipStream_t stream)
{
    dim3 grid(Nc / 16, (M + 63) / 64, batch);
    wmma_gemm_kernel<TB><<<grid, 128, 0, stream>>>(A, B, bias, C, M, Nc, K,
                                                   lda, ldb, ldc, sA, sB, sC, epi);
}

extern "C" void kernel_launch(void* const* d_in, const int* in_sizes, int n_in,
                              void* d_out, int out_size, void* d_ws, size_t ws_size,
                              hipStream_t stream)
{
    const int Bn = 16, N = 512, L = 1024, C = 128, G = 64, H = 4, LAT = 128;

    const int*   atoms      = (const int*)d_in[0];
    const float* atoms_mask = (const float*)d_in[1];
    const int*   adjacency  = (const int*)d_in[2];
    const int*   amino      = (const int*)d_in[3];
    const float* amino_mask = (const float*)d_in[4];
    const float* E_atom     = (const float*)d_in[5];
    const float* E_amino    = (const float*)d_in[6];
    const float* W_gat      = (const float*)d_in[7];   // (H,C,G)
    const float* a_gat      = (const float*)d_in[8];   // (H,2G)
    const float* W_go       = (const float*)d_in[9];   // (G*H,C)
    const float* a_go       = (const float*)d_in[10];  // (2C)
    const float* W_comp_w   = (const float*)d_in[11];  // (LAT,C)
    const float* W_comp_b   = (const float*)d_in[12];
    const float* conv_w     = (const float*)d_in[13];  // (3,1,1,11,11)
    const float* conv_b     = (const float*)d_in[14];  // (3,1)
    const float* W_att_w    = (const float*)d_in[15];  // (LAT,LAT)
    const float* W_att_b    = (const float*)d_in[16];
    const float* pred_w     = (const float*)d_in[17];  // (1,2LAT)
    const float* pred_b     = (const float*)d_in[18];
    float* out = (float*)d_out;
    (void)in_sizes; (void)n_in; (void)out_size; (void)ws_size;

    // workspace bump allocation (floats)
    float* wsf = (float*)d_ws;
    size_t off = 0;
    auto alloc = [&](size_t n) { float* p = wsf + off; off += n; return p; };
    float* av        = alloc((size_t)Bn * N * C);      // atom embeddings (reused as a_v)
    float* whbuf     = alloc((size_t)Bn * N * C);      // per-layer Wh (max width 128)
    float* srcb      = alloc((size_t)Bn * N);
    float* dstb      = alloc((size_t)Bn * N);
    float* att       = alloc((size_t)Bn * N * N);      // attention weights
    float* multi     = alloc((size_t)Bn * N * G * H);  // concat of 4 heads
    float* xbuf      = alloc((size_t)Bn * N * C);
    float* atoms_vec = alloc((size_t)Bn * N * LAT);
    float* pv0       = alloc((size_t)Bn * L * C);      // reused as p_v
    float* pv1       = alloc((size_t)Bn * L * C);
    float* cp        = alloc((size_t)Bn * 2 * LAT);
    float* a_vb = av;
    float* p_vb = pv0;

    const int rowsA = Bn * N;   // 8192
    const int rowsP = Bn * L;   // 16384

    // atom embedding gather
    { long tot = (long)Bn * N * C;
      gather_kernel<<<(int)((tot + 255) / 256), 256, 0, stream>>>(atoms, E_atom, av, tot, C); }

    // --- GAT layer 1: 4 heads, concat ---
    for (int h = 0; h < H; ++h) {
        // Wh = av @ W_gat[h]         (8192x128)(128x64)
        launch_gemm<0>(av, W_gat + (size_t)h * C * G, nullptr, whbuf,
                       rowsA, G, C, C, G, G, 0, 0, 0, 1, 0, stream);
        srcdst_kernel<<<(rowsA + 255) / 256, 256, 0, stream>>>(
            whbuf, a_gat + (size_t)h * 2 * G, G, rowsA, srcb, dstb);
        attn_softmax_kernel<<<rowsA, 256, 0, stream>>>(srcb, dstb, adjacency, att, N);
        // hp = att @ Wh (batched 512x512 @ 512x64), ELU, into multi[:, h*64:(h+1)*64]
        launch_gemm<0>(att, whbuf, nullptr, multi + (size_t)h * G,
                       N, G, N, N, G, G * H,
                       (long)N * N, (long)N * G, (long)N * G * H, Bn, 2, stream);
    }

    // --- GAT layer 2: 256 -> 128, no concat, x = elu(hp) ---
    launch_gemm<0>(multi, W_go, nullptr, whbuf, rowsA, C, G * H, G * H, C, C,
                   0, 0, 0, 1, 0, stream);
    srcdst_kernel<<<(rowsA + 255) / 256, 256, 0, stream>>>(whbuf, a_go, C, rowsA, srcb, dstb);
    attn_softmax_kernel<<<rowsA, 256, 0, stream>>>(srcb, dstb, adjacency, att, N);
    launch_gemm<0>(att, whbuf, nullptr, xbuf, N, C, N, N, C, C,
                   (long)N * N, (long)N * C, (long)N * C, Bn, 2, stream);

    // atoms_vec = leaky(x @ W_comp_w^T + b)
    launch_gemm<1>(xbuf, W_comp_w, W_comp_b, atoms_vec, rowsA, LAT, C,
                   C, C, LAT, 0, 0, 0, 1, 1, stream);

    // --- protein path: gather + 3x (11x11 conv, relu) ---
    { long tot = (long)Bn * L * C;
      gather_kernel<<<(int)((tot + 255) / 256), 256, 0, stream>>>(amino, E_amino, pv0, tot, C);
      const int blk = (int)((tot + 255) / 256);
      conv11_kernel<<<blk, 256, 0, stream>>>(pv0, conv_w +   0, conv_b + 0, pv1, L, C, Bn);
      conv11_kernel<<<blk, 256, 0, stream>>>(pv1, conv_w + 121, conv_b + 1, pv0, L, C, Bn);
      conv11_kernel<<<blk, 256, 0, stream>>>(pv0, conv_w + 242, conv_b + 2, pv1, L, C, Bn);
    }

    // attention projections (leaky(x @ W_att^T + b))
    launch_gemm<1>(atoms_vec, W_att_w, W_att_b, a_vb, rowsA, LAT, LAT,
                   LAT, LAT, LAT, 0, 0, 0, 1, 1, stream);
    launch_gemm<1>(pv1, W_att_w, W_att_b, p_vb, rowsP, LAT, LAT,
                   LAT, LAT, LAT, 0, 0, 0, 1, 1, stream);

    // pooled concat + leaky x2, then prediction head
    pool_cp_kernel<<<Bn, 2 * LAT, 0, stream>>>(a_vb, p_vb, atoms_mask, amino_mask, cp, N, L, LAT);
    pred_kernel<<<Bn, 2 * LAT, 0, stream>>>(cp, pred_w, pred_b, out, 2 * LAT);
}
